// PerHeadSparseMoE_85160611545785
// MI455X (gfx1250) — matrix operand order repaired and checked by reading, not compiled, not run on hardware
//
#include <hip/hip_runtime.h>

typedef _Float16 f16;
typedef __attribute__((ext_vector_type(16))) _Float16 v16h;
typedef __attribute__((ext_vector_type(8)))  _Float16 v8h;
typedef __attribute__((ext_vector_type(4)))  _Float16 v4h;
typedef __attribute__((ext_vector_type(8)))  float    v8f;

#define BDIM 128
#define HDIM 384
#define NHEAD 16
#define NEXP 8
#define TOKTILE 128          // tokens per block in main kernel
#define XS_STRIDE 128        // no pad: A-frags read once into registers
#define W1_STRIDE 136        // 128 + 8 pad (halves), [h][d]
#define W2_STRIDE 40         // 32 + 8 pad (halves), [d][h]
#define HS_STRIDE 40         // 32 + 8 pad (halves)
#define USE_ASYNC_LDS 1      // gfx1250 async global->LDS copies for weight slabs

// ---------------- async global->LDS b128 copy (gfx1250, ASYNCcnt) -------------
// GVS mode: saddr = uniform 64-bit base (SGPR pair), vaddr = per-lane 32-bit
// byte offset, vdst = per-lane LDS byte address. Tracked by ASYNCcnt.
__device__ __forceinline__ void seg_copy_b128(f16* dst_lds, const f16* src_base,
                                              unsigned byte_off) {
#if USE_ASYNC_LDS
  unsigned lds_addr = (unsigned)(uintptr_t)dst_lds;              // addr[31:0] = LDS offset
  unsigned long long gbase = (unsigned long long)(uintptr_t)src_base;
  asm volatile("global_load_async_to_lds_b128 %0, %1, %2"
               :: "v"(lds_addr), "v"(byte_off), "s"(gbase)
               : "memory");
#else
  *(v8h*)dst_lds = *(const v8h*)((const char*)src_base + byte_off);
#endif
}
__device__ __forceinline__ void wait_async_copies() {
#if USE_ASYNC_LDS
  asm volatile("s_wait_asynccnt 0x0" ::: "memory");
#endif
}

// fast silu: g * sigmoid(g) with 1-ULP v_rcp_f32 (f16 pipeline dominates error)
__device__ __forceinline__ float fast_silu(float g) {
  return g * __builtin_amdgcn_rcpf(1.0f + __expf(-g));
}

// ---------------- fragment loaders (gfx1250 wave32 WMMA f16 layouts) ----------
// A 16x32 f16: lane&15 -> M, K = 2*kb + i (16 contiguous), kb = (lane>>4)*8
__device__ __forceinline__ v16h afrag(const f16* base, int stride, int k0, int m, int kb) {
  v16h a;
  const f16* p = base + m * stride + k0 + 2 * kb;
#pragma unroll
  for (int i = 0; i < 16; ++i) a[i] = p[i];
  return a;
}
// B 32x16 f16 from an LDS tile stored transposed [n][k]:
// lane&15 -> N, K = kb2 + i, kb2 = (lane>>4)*16 -> 16 contiguous halves
__device__ __forceinline__ v16h bfrag(const f16* T, int stride, int n0, int k0, int nl, int kb2) {
  v16h b;
  const f16* p = T + (n0 + nl) * stride + k0 + kb2;
#pragma unroll
  for (int i = 0; i < 16; ++i) b[i] = p[i];
  return b;
}

// ---------------- kernel 0: transpose + convert weights to f16 ----------------
// src: EN matrices of R x C (f32, row-major) -> dst: EN matrices of C x R (f16)
__global__ __launch_bounds__(256) void transpose_cvt_kernel(
    const float* __restrict__ src, f16* __restrict__ dst,
    int R, int C, int tilesR, int tilesC) {
  __shared__ float t[32][33];
  const int tid = threadIdx.x;
  const int b   = blockIdx.x;
  const int mat = b / (tilesR * tilesC);
  const int rem = b % (tilesR * tilesC);
  const int tr  = rem / tilesC;
  const int tc  = rem % tilesC;
  const float* s = src + (size_t)mat * R * C + (size_t)(tr * 32) * C + tc * 32;
  f16*         d = dst + (size_t)mat * R * C + (size_t)(tc * 32) * R + tr * 32;
  const int col = tid & 31;
#pragma unroll
  for (int q = 0; q < 4; ++q) {
    int r = (tid >> 5) + q * 8;
    t[r][col] = s[(size_t)r * C + col];
  }
  __syncthreads();
#pragma unroll
  for (int q = 0; q < 4; ++q) {
    int r = (tid >> 5) + q * 8;          // dst row (= src col)
    d[(size_t)r * R + col] = (f16)t[col][r];
  }
}

// ---------------- kernel 1: router (logits, top-2, softmax) -------------------
__global__ __launch_bounds__(256) void router_kernel(
    const float* __restrict__ x, const float* __restrict__ rw,
    int* __restrict__ selE, float* __restrict__ selW) {
  const int b = blockIdx.x;
  const int t = threadIdx.x;
  const float* xb = x + (size_t)b * (NHEAD * BDIM);
  float p[NEXP];
#pragma unroll
  for (int e = 0; e < NEXP; ++e) p[e] = 0.0f;
  for (int i = t * 4; i < NHEAD * BDIM; i += 1024) {
    const float4 xv = *(const float4*)(xb + i);
#pragma unroll
    for (int e = 0; e < NEXP; ++e) {
      const float4 rv = *(const float4*)(rw + e * (NHEAD * BDIM) + i);
      p[e] += xv.x * rv.x + xv.y * rv.y + xv.z * rv.z + xv.w * rv.w;
    }
  }
  __shared__ float red[NEXP][256];
#pragma unroll
  for (int e = 0; e < NEXP; ++e) red[e][t] = p[e];
  __syncthreads();
  for (int s = 128; s > 0; s >>= 1) {
    if (t < s) {
#pragma unroll
      for (int e = 0; e < NEXP; ++e) red[e][t] += red[e][t + s];
    }
    __syncthreads();
  }
  if (t == 0) {
    float l[NEXP];
#pragma unroll
    for (int e = 0; e < NEXP; ++e) l[e] = red[e][0];
    int e0 = 0; float b0 = l[0];
#pragma unroll
    for (int e = 1; e < NEXP; ++e) if (l[e] > b0) { b0 = l[e]; e0 = e; }
    int e1 = (e0 == 0) ? 1 : 0; float b1 = l[e1];
#pragma unroll
    for (int e = 0; e < NEXP; ++e)
      if (e != e0 && e != ((e0 == 0) ? 1 : 0) && l[e] > b1) { b1 = l[e]; e1 = e; }
    float x1 = __expf(b1 - b0);
    float s = 1.0f + x1;
    selE[2 * b]     = e0;  selW[2 * b]     = 1.0f / s;
    selE[2 * b + 1] = e1;  selW[2 * b + 1] = x1 / s;
  }
}

// ---------------- kernel 2: deterministic per-expert compaction ---------------
__global__ __launch_bounds__(256) void compact_kernel(
    const int* __restrict__ selE, const float* __restrict__ selW,
    int* __restrict__ listTok, float* __restrict__ listW, int* __restrict__ count,
    int B) {
  const int e = blockIdx.x;
  const int t = threadIdx.x;
  __shared__ int sc[256];
  int base = 0;
  for (int c = 0; c < B; c += 256) {
    int b = c + t;
    int pr = 0; float w = 0.0f;
    int ea = selE[2 * b], eb = selE[2 * b + 1];
    if (ea == e)      { pr = 1; w = selW[2 * b]; }
    else if (eb == e) { pr = 1; w = selW[2 * b + 1]; }
    sc[t] = pr;
    __syncthreads();
    for (int off = 1; off < 256; off <<= 1) {
      int v = (t >= off) ? sc[t - off] : 0;
      __syncthreads();
      sc[t] += v;
      __syncthreads();
    }
    int incl = sc[t];
    int tot  = sc[255];
    if (pr) {
      int pos = base + incl - 1;
      listTok[e * B + pos] = b;
      listW [e * B + pos] = w;
    }
    base += tot;
    __syncthreads();
  }
  if (t == 0) count[e] = base;
}

// ---------------- kernel 3: main MoE WMMA kernel (double-buffered async) ------
__global__ __launch_bounds__(256) void moe_kernel(
    const float* __restrict__ x,
    const f16* __restrict__ w1T,   // [e][n][h][d] f16
    const f16* __restrict__ wgT,   // [e][n][h][d] f16
    const f16* __restrict__ w2T,   // [e][n][d][h] f16
    const int* __restrict__ listTok, const float* __restrict__ listW,
    const int* __restrict__ count,
    float* __restrict__ out, int B, int maxTiles) {
  const int e    = blockIdx.x / maxTiles;
  const int tile = blockIdx.x % maxTiles;
  const int n    = blockIdx.y;
  const int cnt  = count[e];
  if (tile * TOKTILE >= cnt) return;

  const int tid  = threadIdx.x;
  const int wave = tid >> 5;
  const int lane = tid & 31;
  const int nl   = lane & 15;
  const int kb   = (lane >> 4) * 8;   // A-frag K base
  const int kb2  = (lane >> 4) * 16;  // B-frag K base

  // double-buffered weight slabs (gfx1250 big LDS: 99,328 B total)
  __shared__ __align__(16) f16  w1s[2 * 32 * W1_STRIDE];    // 17408 B
  __shared__ __align__(16) f16  wgs[2 * 32 * W1_STRIDE];    // 17408 B
  __shared__ __align__(16) f16  w2s[2 * BDIM * W2_STRIDE];  // 20480 B
  __shared__ __align__(16) f16  hact[8 * 16 * HS_STRIDE];   // 10240 B
  __shared__ __align__(16) f16  xs[TOKTILE * XS_STRIDE];    // 32768 B
  __shared__ int   tokL[TOKTILE];
  __shared__ float wL[TOKTILE];

  const f16* w1e = w1T + (size_t)(e * NHEAD + n) * HDIM * BDIM;  // [h][d]
  const f16* wge = wgT + (size_t)(e * NHEAD + n) * HDIM * BDIM;  // [h][d]
  const f16* w2e = w2T + (size_t)(e * NHEAD + n) * BDIM * HDIM;  // [d][h]

  // issue all three weight slabs for slab hp into buffer set `buf`
  auto issue_slabs = [&](int hp, int buf) {
    const f16* w1slab = w1e + (size_t)(hp * 32) * BDIM;
    const f16* wgslab = wge + (size_t)(hp * 32) * BDIM;
    const f16* w2slab = w2e + hp * 32;
    f16* w1d = w1s + buf * (32 * W1_STRIDE);
    f16* wgd = wgs + buf * (32 * W1_STRIDE);
    f16* w2d = w2s + buf * (BDIM * W2_STRIDE);
#pragma unroll
    for (int q = 0; q < 2; ++q) {
      int s  = q * 256 + tid;           // [0,512) 8-half segments
      int hh = s >> 4, dd = (s & 15) * 8;
      unsigned off = (unsigned)((hh * BDIM + dd) * sizeof(f16));
      seg_copy_b128(&w1d[hh * W1_STRIDE + dd], w1slab, off);
      seg_copy_b128(&wgd[hh * W1_STRIDE + dd], wgslab, off);
    }
#pragma unroll
    for (int q = 0; q < 2; ++q) {
      int s = q * 256 + tid;
      int d = s >> 2, hh = (s & 3) * 8;
      seg_copy_b128(&w2d[d * W2_STRIDE + hh], w2slab,
                    (unsigned)((d * HDIM + hh) * sizeof(f16)));
    }
  };

  // ---- gather token list for this tile ----
  if (tid < TOKTILE) {
    int j = tile * TOKTILE + tid;
    if (j < cnt) { tokL[tid] = listTok[e * B + j]; wL[tid] = listW[e * B + j]; }
    else         { tokL[tid] = 0;                  wL[tid] = 0.0f; }
  }

  // prologue: prefetch slab 0 into buffer 0 (overlaps with X gather below)
  issue_slabs(0, 0);
  __syncthreads();  // tokL visible

  // ---- gather X tile (128 tokens x 128) -> f16 LDS ----
#pragma unroll
  for (int q = 0; q < 16; ++q) {
    int fidx = q * 256 + tid;          // [0, 4096) float4 slots
    int row  = fidx >> 5;              // 32 float4 per row
    int c4   = (fidx & 31) * 4;
    const float4 xv = *(const float4*)(x + (size_t)tokL[row] * (NHEAD * BDIM) + n * BDIM + c4);
    v4h h4 = { (f16)xv.x, (f16)xv.y, (f16)xv.z, (f16)xv.w };
    *(v4h*)(xs + row * XS_STRIDE + c4) = h4;
  }
  __syncthreads();

  // ---- preload invariant X A-fragments (16 tokens x 128) into registers ----
  const f16* xrow = xs + (wave * 16) * XS_STRIDE;
  v16h xA[4];
#pragma unroll
  for (int kk = 0; kk < 4; ++kk) xA[kk] = afrag(xrow, XS_STRIDE, kk * 32, nl, kb);

  v8f outC[8];
#pragma unroll
  for (int dt = 0; dt < 8; ++dt) outC[dt] = (v8f)(0.0f);

  f16* hs = hact + wave * 16 * HS_STRIDE;

  // ---- pipelined loop over H in 32-column slabs: 1 wait + 1 barrier / iter ----
  int cur = 0;
  for (int hp = 0; hp < HDIM / 32; ++hp) {
    wait_async_copies();   // slab(hp) resident (its fetch overlapped compute(hp-1))
    __syncthreads();       // all waves' reads of buf[cur^1] (compute hp-1) done too

    if (hp + 1 < HDIM / 32) issue_slabs(hp + 1, cur ^ 1);  // overlaps compute(hp)

    const f16* w1cur = w1s + cur * (32 * W1_STRIDE);
    const f16* wgcur = wgs + cur * (32 * W1_STRIDE);
    const f16* w2cur = w2s + cur * (BDIM * W2_STRIDE);

    // H1 = X @ W1[:, slab] (two 16x16 tiles, K=128 in 4 steps)
    v8f c0 = (v8f)(0.0f), c1 = (v8f)(0.0f);
#pragma unroll
    for (int kk = 0; kk < 4; ++kk) {
      v16h b0 = bfrag(w1cur, W1_STRIDE, 0,  kk * 32, nl, kb2);
      v16h b1 = bfrag(w1cur, W1_STRIDE, 16, kk * 32, nl, kb2);
      c0 = __builtin_amdgcn_wmma_f32_16x16x32_f16(false, xA[kk], false, b0, (short)0, c0, false, false);
      c1 = __builtin_amdgcn_wmma_f32_16x16x32_f16(false, xA[kk], false, b1, (short)0, c1, false, false);
    }
    // G = X @ Wg[:, slab]
    v8f g0 = (v8f)(0.0f), g1 = (v8f)(0.0f);
#pragma unroll
    for (int kk = 0; kk < 4; ++kk) {
      v16h b0 = bfrag(wgcur, W1_STRIDE, 0,  kk * 32, nl, kb2);
      v16h b1 = bfrag(wgcur, W1_STRIDE, 16, kk * 32, nl, kb2);
      g0 = __builtin_amdgcn_wmma_f32_16x16x32_f16(false, xA[kk], false, b0, (short)0, g0, false, false);
      g1 = __builtin_amdgcn_wmma_f32_16x16x32_f16(false, xA[kk], false, b1, (short)0, g1, false, false);
    }

    // SwiGLU in f32 (v_exp + v_rcp), store activation tile (16 x 32) as [m][h]
#pragma unroll
    for (int v = 0; v < 8; ++v) {
      int m = v + 8 * (lane >> 4);
      hs[m * HS_STRIDE + nl]      = (f16)(c0[v] * fast_silu(g0[v]));
      hs[m * HS_STRIDE + 16 + nl] = (f16)(c1[v] * fast_silu(g1[v]));
    }
    // within-wave LDS RAW: hardware keeps DS ops in order per wave

    // out(16x128) += Hact(16x32) @ W2slab(32x128)
    v16h a2 = afrag(hs, HS_STRIDE, 0, nl, kb);
#pragma unroll
    for (int dt = 0; dt < 8; ++dt) {
      v16h b2 = bfrag(w2cur, W2_STRIDE, dt * 16, 0, nl, kb2);
      outC[dt] = __builtin_amdgcn_wmma_f32_16x16x32_f16(false, a2, false, b2, (short)0, outC[dt], false, false);
    }
    cur ^= 1;
  }

  // ---- scale rows by routing weight, accumulate (2 commutative adds/elem) ----
#pragma unroll
  for (int v = 0; v < 8; ++v) {
    int m   = v + 8 * (lane >> 4);
    int row = wave * 16 + m;
    float w = wL[row];
    if (w != 0.0f) {
      float* ob = out + (size_t)tokL[row] * (NHEAD * BDIM) + n * BDIM + nl;
#pragma unroll
      for (int dt = 0; dt < 8; ++dt) {
        atomicAdd(ob + dt * 16, outC[dt][v] * w);
      }
    }
  }
}

// ---------------- launcher ----------------------------------------------------
extern "C" void kernel_launch(void* const* d_in, const int* in_sizes, int n_in,
                              void* d_out, int out_size, void* d_ws, size_t ws_size,
                              hipStream_t stream) {
  const float* x   = (const float*)d_in[0];
  const float* rw  = (const float*)d_in[1];
  const float* w1  = (const float*)d_in[2];
  const float* wg  = (const float*)d_in[3];
  const float* w2  = (const float*)d_in[4];
  float* out = (float*)d_out;

  const int B = in_sizes[0] / (NHEAD * BDIM);
  const size_t wElems = (size_t)NEXP * NHEAD * BDIM * HDIM;

  // workspace layout
  char* ws = (char*)d_ws;
  f16*   w1T     = (f16*)ws;   ws += wElems * sizeof(f16);
  f16*   wgT     = (f16*)ws;   ws += wElems * sizeof(f16);
  f16*   w2T     = (f16*)ws;   ws += wElems * sizeof(f16);
  int*   selE    = (int*)ws;   ws += (size_t)B * 2 * sizeof(int);
  float* selW    = (float*)ws; ws += (size_t)B * 2 * sizeof(float);
  int*   listTok = (int*)ws;   ws += (size_t)NEXP * B * sizeof(int);
  float* listW   = (float*)ws; ws += (size_t)NEXP * B * sizeof(float);
  int*   count   = (int*)ws;   ws += 256;

  hipMemsetAsync(out, 0, (size_t)out_size * sizeof(float), stream);

  // one-shot weight transpose+convert: w1/wg (128x384 -> [h][d]), w2 (384x128 -> [d][h])
  {
    const int EN = NEXP * NHEAD;
    transpose_cvt_kernel<<<EN * (BDIM / 32) * (HDIM / 32), 256, 0, stream>>>(
        w1, w1T, BDIM, HDIM, BDIM / 32, HDIM / 32);
    transpose_cvt_kernel<<<EN * (BDIM / 32) * (HDIM / 32), 256, 0, stream>>>(
        wg, wgT, BDIM, HDIM, BDIM / 32, HDIM / 32);
    transpose_cvt_kernel<<<EN * (HDIM / 32) * (BDIM / 32), 256, 0, stream>>>(
        w2, w2T, HDIM, BDIM, HDIM / 32, BDIM / 32);
  }

  router_kernel<<<B, 256, 0, stream>>>(x, rw, selE, selW);
  compact_kernel<<<NEXP, 256, 0, stream>>>(selE, selW, listTok, listW, count, B);

  const int maxTiles = (B + TOKTILE - 1) / TOKTILE;
  dim3 grid(NEXP * maxTiles, NHEAD);
  moe_kernel<<<grid, 256, 0, stream>>>(x, w1T, wgT, w2T, listTok, listW, count, out, B, maxTiles);
}
